// GATBot_74371653697682
// MI455X (gfx1250) — matrix operand order repaired and checked by reading, not compiled, not run on hardware
//
#include <hip/hip_runtime.h>
#include <hip/hip_bf16.h>

#define N_NODES 10000
#define E_EDGES 320000
#define ET      (E_EDGES + N_NODES)   // self loops appended
#define F_IN    256
#define HID     64
#define HEADS   8
#define NEG_SLOPE 0.2f

typedef __attribute__((ext_vector_type(16))) _Float16 v16h;
typedef __attribute__((ext_vector_type(8)))  _Float16 v8h;
typedef __attribute__((ext_vector_type(8)))  float    v8f;

#define CEIL(a,b) (((a)+(b)-1)/(b))

// ---------- order-preserving float <-> uint for atomicMax-based segment max ----
__device__ __forceinline__ unsigned fenc(float f) {
    unsigned u = __float_as_uint(f);
    return (u & 0x80000000u) ? ~u : (u | 0x80000000u);
}
__device__ __forceinline__ float fdec(unsigned o) {
    unsigned u = (o & 0x80000000u) ? (o & 0x7fffffffu) : ~o;
    return __uint_as_float(u);
}

// ---------- edge list build: int64 -> int32, append self loops ----------------
__global__ void k_build_edges(const long long* __restrict__ ei,
                              int* __restrict__ src, int* __restrict__ dst) {
    int i = blockIdx.x * blockDim.x + threadIdx.x;
    if (i < E_EDGES) {
        src[i] = (int)ei[i];
        dst[i] = (int)ei[E_EDGES + i];
    } else if (i < ET) {
        int n = i - E_EDGES;
        src[i] = n; dst[i] = n;
    }
}

// ---------- f32 -> f16 elementwise --------------------------------------------
__global__ void k_f32_to_f16(const float* __restrict__ in, _Float16* __restrict__ out, int n) {
    int i = blockIdx.x * blockDim.x + threadIdx.x;
    if (i < n) out[i] = (_Float16)in[i];
}

// ---------- weight convert + transpose: W[K x Nout] f32 -> Wt[Nout x K] f16 ---
__global__ void k_w_transpose_f16(const float* __restrict__ W, _Float16* __restrict__ Wt,
                                  int K, int Nout) {
    int t = blockIdx.x * blockDim.x + threadIdx.x;
    if (t < K * Nout) {
        int kk = t / Nout, nn = t - kk * Nout;
        Wt[(size_t)nn * K + kk] = (_Float16)W[t];
    }
}

// ---------- WMMA GEMM: C[M x Nout] = A[M x K] * Bt[Nout x K]^T ----------------
// f16 inputs, f32 accumulate. Each wave computes a 16x64 output strip
// (4 n-tiles) so the A fragment is loaded once and reused by 4 WMMAs per
// k-step. grid = (M/16, CEIL(Nout,256)), block = 128 (4 waves).
__global__ void __launch_bounds__(128) k_gemm_wmma(const _Float16* __restrict__ A,
                                                   const _Float16* __restrict__ Bt,
                                                   float* __restrict__ C,
                                                   int K, int Nout) {
    const int lane = threadIdx.x & 31;
    const int wave = threadIdx.x >> 5;
    const int tm   = blockIdx.x;
    const int tn0  = (blockIdx.y * 4 + wave) * 4;   // first of 4 n-tiles
    if (tn0 * 16 >= Nout) return;                   // wave-uniform: EXEC all-1s
    const int hi  = lane >> 4;                      // half-wave select
    const int l15 = lane & 15;
    const int row = tm * 16 + l15;                  // A-frag row for this lane
    const _Float16* __restrict__ Ar = A  + (size_t)row * K;
    const _Float16* __restrict__ Br = Bt + (size_t)(tn0 * 16 + l15) * K;
    const size_t bstride = (size_t)16 * K;          // halves between n-tiles

    v8f acc0 = {}, acc1 = {}, acc2 = {}, acc3 = {};
    for (int k = 0; k < K; k += 32) {
        // A 16x32 f16 frag: halves 0..15 = K ka..ka+15 (ka = k + hi*16)
        const int ka = k + hi * 16;
        v8h a_lo = *(const v8h*)(Ar + ka);
        v8h a_hi = *(const v8h*)(Ar + ka + 8);
        v16h a = __builtin_shufflevector(a_lo, a_hi,
                                         0,1,2,3,4,5,6,7,8,9,10,11,12,13,14,15);
        // B 32x16 f16 frags: lanes 0-15 hold K=k..k+15, lanes 16-31 K=k+16..k+31
        const int kb = k + hi * 16;
        v16h b0 = *(const v16h*)(Br + kb);
        v16h b1 = *(const v16h*)(Br + bstride + kb);
        v16h b2 = *(const v16h*)(Br + 2 * bstride + kb);
        v16h b3 = *(const v16h*)(Br + 3 * bstride + kb);
        acc0 = __builtin_amdgcn_wmma_f32_16x16x32_f16(false, a, false, b0, (short)0, acc0, false, false);
        acc1 = __builtin_amdgcn_wmma_f32_16x16x32_f16(false, a, false, b1, (short)0, acc1, false, false);
        acc2 = __builtin_amdgcn_wmma_f32_16x16x32_f16(false, a, false, b2, (short)0, acc2, false, false);
        acc3 = __builtin_amdgcn_wmma_f32_16x16x32_f16(false, a, false, b3, (short)0, acc3, false, false);
    }
    // C layout: VGPR i -> M = i + hi*8, N = l15 (per 16x16 tile)
    const int r0 = tm * 16 + hi * 8;
    float* __restrict__ Cout = C + (size_t)r0 * Nout + tn0 * 16 + l15;
#pragma unroll
    for (int i = 0; i < 8; i++) {
        float* cp = Cout + (size_t)i * Nout;
        cp[0]  = acc0[i];
        cp[16] = acc1[i];
        cp[32] = acc2[i];
        cp[48] = acc3[i];
    }
}

// ---------- per-node attention coefficients ----------------------------------
__global__ void k_alphas(const float* __restrict__ h, const float* __restrict__ a_src,
                         const float* __restrict__ a_dst, float* __restrict__ as,
                         float* __restrict__ ad, int H) {
    int t = blockIdx.x * blockDim.x + threadIdx.x;
    if (t >= N_NODES * H) return;
    int n = t / H, hd = t - n * H;
    const float* hp  = h + (size_t)n * H * HID + hd * HID;
    const float* asr = a_src + hd * HID;
    const float* adr = a_dst + hd * HID;
    float s = 0.f, d = 0.f;
#pragma unroll 8
    for (int c = 0; c < HID; c++) { float v = hp[c]; s += v * asr[c]; d += v * adr[c]; }
    as[t] = s; ad[t] = d;
}

// ---------- edge pass 1: segment max of leaky-relu logits --------------------
__global__ void k_edge_max(const int* __restrict__ src, const int* __restrict__ dst,
                           const float* __restrict__ as, const float* __restrict__ ad,
                           unsigned* __restrict__ m_ord, int H) {
    int t = blockIdx.x * blockDim.x + threadIdx.x;
    if (t >= ET * H) return;
    int e = t / H, hd = t - e * H;
    int s = src[e], d = dst[e];
    float z = as[s * H + hd] + ad[d * H + hd];
    z = z > 0.f ? z : NEG_SLOPE * z;
    atomicMax(&m_ord[d * H + hd], fenc(z));
}

// ---------- edge pass 2: denom += ex ; agg += ex * h[src] --------------------
__global__ void k_edge_agg(const int* __restrict__ src, const int* __restrict__ dst,
                           const float* __restrict__ as, const float* __restrict__ ad,
                           const unsigned* __restrict__ m_ord, float* __restrict__ denom,
                           const float* __restrict__ h, float* __restrict__ agg, int H) {
    int t = blockIdx.x * blockDim.x + threadIdx.x;
    if (t >= ET * H) return;
    int e = t / H, hd = t - e * H;
    int s = src[e], d = dst[e];
    float z = as[s * H + hd] + ad[d * H + hd];
    z = z > 0.f ? z : NEG_SLOPE * z;
    float ex = expf(z - fdec(m_ord[d * H + hd]));
    atomicAdd(&denom[d * H + hd], ex);
    const float* __restrict__ hp = h   + (size_t)s * H * HID + hd * HID;
    float*       __restrict__ ap = agg + (size_t)d * H * HID + hd * HID;
#pragma unroll 4
    for (int c = 0; c < HID; c++) atomicAdd(&ap[c], ex * hp[c]);
}

// ---------- finalize: normalize, + bias, optional ELU, emit f16 and/or f32 ---
__global__ void k_finalize(const float* __restrict__ agg, const float* __restrict__ denom,
                           const float* __restrict__ bias, _Float16* __restrict__ out16,
                           float* __restrict__ out32, int H, int do_elu) {
    int t = blockIdx.x * blockDim.x + threadIdx.x;
    int hc = H * HID;
    if (t >= N_NODES * hc) return;
    int n = t / hc; int rem = t - n * hc; int hd = rem / HID;
    float v = agg[t] / (denom[n * H + hd] + 1e-16f) + bias[rem];
    if (do_elu) v = v > 0.f ? v : (expf(v) - 1.f);
    if (out16) out16[t] = (_Float16)v;
    if (out32) out32[t] = v;
}

// ---------- classifier + log_softmax (2 classes) -----------------------------
__global__ void k_classifier(const float* __restrict__ h, const float* __restrict__ Wl,
                             const float* __restrict__ bl, float* __restrict__ out) {
    int n = blockIdx.x * blockDim.x + threadIdx.x;
    if (n >= N_NODES) return;
    const float* hp = h + (size_t)n * HID;
    float l0 = bl[0], l1 = bl[1];
#pragma unroll 8
    for (int c = 0; c < HID; c++) { float v = hp[c]; l0 += v * Wl[c * 2]; l1 += v * Wl[c * 2 + 1]; }
    float mx  = fmaxf(l0, l1);
    float lse = mx + logf(expf(l0 - mx) + expf(l1 - mx));
    out[n * 2]     = l0 - lse;
    out[n * 2 + 1] = l1 - lse;
}

extern "C" void kernel_launch(void* const* d_in, const int* in_sizes, int n_in,
                              void* d_out, int out_size, void* d_ws, size_t ws_size,
                              hipStream_t stream) {
    (void)in_sizes; (void)n_in; (void)out_size; (void)ws_size;
    const float*     x    = (const float*)d_in[0];
    const long long* ei   = (const long long*)d_in[1];
    const float*     W1   = (const float*)d_in[2];
    const float*     as1  = (const float*)d_in[3];
    const float*     ad1  = (const float*)d_in[4];
    const float*     b1   = (const float*)d_in[5];
    const float*     W2   = (const float*)d_in[6];
    const float*     as2  = (const float*)d_in[7];
    const float*     ad2  = (const float*)d_in[8];
    const float*     b2   = (const float*)d_in[9];
    const float*     W3   = (const float*)d_in[10];
    const float*     as3  = (const float*)d_in[11];
    const float*     ad3  = (const float*)d_in[12];
    const float*     b3   = (const float*)d_in[13];
    const float*     linW = (const float*)d_in[14];
    const float*     linb = (const float*)d_in[15];

    // ---- workspace carve-up (256B aligned) ----
    char* ws = (char*)d_ws;
    size_t off = 0;
    auto alloc = [&](size_t bytes) -> char* {
        char* p = ws + off;
        off = (off + bytes + 255) & ~(size_t)255;
        return p;
    };
    int*      src32 = (int*)alloc((size_t)ET * 4);
    int*      dst32 = (int*)alloc((size_t)ET * 4);
    _Float16* act16 = (_Float16*)alloc((size_t)N_NODES * 512 * 2);  // f16 activations
    _Float16* wt16  = (_Float16*)alloc((size_t)512 * 512 * 2);      // f16 transposed W
    float*    hproj = (float*)alloc((size_t)N_NODES * 512 * 4);     // GEMM output
    float*    asbuf = (float*)alloc((size_t)N_NODES * HEADS * 4);
    float*    adbuf = (float*)alloc((size_t)N_NODES * HEADS * 4);
    unsigned* mord  = (unsigned*)alloc((size_t)N_NODES * HEADS * 4);
    float*    denom = (float*)alloc((size_t)N_NODES * HEADS * 4);
    float*    agg   = (float*)alloc((size_t)N_NODES * 512 * 4);
    float*    h3    = (float*)alloc((size_t)N_NODES * HID * 4);

    k_build_edges<<<CEIL(ET, 256), 256, 0, stream>>>(ei, src32, dst32);
    k_f32_to_f16<<<CEIL(N_NODES * F_IN, 256), 256, 0, stream>>>(x, act16, N_NODES * F_IN);

    auto gat_layer = [&](const _Float16* in16, int K, const float* W, const float* asr,
                         const float* adr, const float* b, int H,
                         _Float16* o16, float* o32, int elu) {
        const int Nout = H * HID;
        k_w_transpose_f16<<<CEIL(K * Nout, 256), 256, 0, stream>>>(W, wt16, K, Nout);
        dim3 gg(N_NODES / 16, CEIL(Nout, 256));   // 4 waves x 4 n-tiles = 256 cols/block
        k_gemm_wmma<<<gg, 128, 0, stream>>>(in16, wt16, hproj, K, Nout);
        k_alphas<<<CEIL(N_NODES * H, 256), 256, 0, stream>>>(hproj, asr, adr, asbuf, adbuf, H);
        hipMemsetAsync(mord,  0, (size_t)N_NODES * H * 4, stream);
        hipMemsetAsync(denom, 0, (size_t)N_NODES * H * 4, stream);
        hipMemsetAsync(agg,   0, (size_t)N_NODES * Nout * 4, stream);
        k_edge_max<<<CEIL(ET * H, 256), 256, 0, stream>>>(src32, dst32, asbuf, adbuf, mord, H);
        k_edge_agg<<<CEIL(ET * H, 256), 256, 0, stream>>>(src32, dst32, asbuf, adbuf, mord,
                                                          denom, hproj, agg, H);
        k_finalize<<<CEIL(N_NODES * Nout, 256), 256, 0, stream>>>(agg, denom, b, o16, o32, H, elu);
    };

    // layer 1: 256 -> 8x64 concat, elu
    gat_layer(act16, F_IN, W1, as1, ad1, b1, HEADS, act16, nullptr, 1);
    // layer 2: 512 -> 8x64 concat, elu
    gat_layer(act16, HEADS * HID, W2, as2, ad2, b2, HEADS, act16, nullptr, 1);
    // layer 3: 512 -> 1x64 (mean over 1 head == identity), no elu, keep f32
    gat_layer(act16, HEADS * HID, W3, as3, ad3, b3, 1, nullptr, h3, 0);

    k_classifier<<<CEIL(N_NODES, 256), 256, 0, stream>>>(h3, linW, linb, (float*)d_out);
}
